// MNISTExpertPOC_49718541418736
// MI455X (gfx1250) — matrix operand order repaired and checked by reading, not compiled, not run on hardware
//
#include <hip/hip_runtime.h>
#include <math.h>

#define BATCH   16384
#define IN_DIM  784
#define KPAD    800          // IN_DIM padded to multiple of 32
#define HID     768
#define NEXP    8
#define NCLS    10
#define DELTA0  7.0f
#define DECAY   0.7f
#define KSEL    32
#define KWIN    5
#define NT_HID  (HID / 16)   // 48 column tiles
#define KC_IN   (KPAD / 32)  // 25 k-chunks for the input GEMM
#define KC_HID  (HID / 32)   // 24 k-chunks for the expert GEMM

typedef __attribute__((ext_vector_type(16))) __bf16 bf16x16;
typedef __attribute__((ext_vector_type(8)))  __bf16 bf16x8;
typedef __attribute__((ext_vector_type(8)))  float  floatx8;
typedef __attribute__((ext_vector_type(4)))  unsigned int u32x4;
typedef __attribute__((ext_vector_type(8)))  int i32x8;
typedef __attribute__((ext_vector_type(4)))  int i32x4;

// ---------------------------------------------------------------------------
// WMMA helpers (CDNA5 wave32, v_wmma_f32_16x16x32_bf16)
// ---------------------------------------------------------------------------
static __device__ __forceinline__ floatx8 wmma_bf16(bf16x16 a, bf16x16 b, floatx8 c) {
  return __builtin_amdgcn_wmma_f32_16x16x32_bf16(false, a, false, b, (short)0, c,
                                                 false, false);
}

static __device__ __forceinline__ bf16x16 cat16(bf16x8 lo, bf16x8 hi) {
  return __builtin_shufflevector(lo, hi, 0, 1, 2, 3, 4, 5, 6, 7,
                                 8, 9, 10, 11, 12, 13, 14, 15);
}

// A fragment straight from a bf16 row-major activation matrix.
static __device__ __forceinline__ bf16x16
load_a_bf(const __bf16* __restrict__ A, int lda, int m0, int k0, int lane) {
  int row = lane & 15;
  int kb  = k0 + ((lane & 16) ? 16 : 0);
  const __bf16* p = A + (size_t)(m0 + row) * lda + kb;
  bf16x8 lo = *(const bf16x8*)p;
  bf16x8 hi = *(const bf16x8*)(p + 8);
  return cat16(lo, hi);
}

// B fragment from a TDM-staged LDS panel (fragment-native order):
// lane's 16 bf16 are contiguous -> two aligned ds_load_b128.
static __device__ __forceinline__ bf16x16
load_b_lds(const __bf16* panel, int c, int lane) {
  const __bf16* p = panel + ((size_t)c * 32 + lane) * 16;
  bf16x8 lo = *(const bf16x8*)p;
  bf16x8 hi = *(const bf16x8*)(p + 8);
  return cat16(lo, hi);
}

// B fragment directly from the swizzled global buffer (small GEMM path).
static __device__ __forceinline__ bf16x16
load_b_sw(const __bf16* __restrict__ Wsw, int NT, int c, int t, int lane) {
  const __bf16* p = Wsw + (((size_t)c * NT + t) * 32 + lane) * 16;
  bf16x8 lo = *(const bf16x8*)p;
  bf16x8 hi = *(const bf16x8*)(p + 8);
  return cat16(lo, hi);
}

// ---------------------------------------------------------------------------
// TDM: async 2D panel copy global->LDS. rows x 1KB tiles, row stride 48KB.
// D# per CDNA5 ISA ch.8: group0 {count, lds_addr, global_addr, type=2},
// group1 {data_size=4B, tensor/tile dims in dwords, dim0 stride}, g2/g3/g4=0.
// This toolchain exposes the 6-arg builtin: (g0, g1, g2, g3, g4, cpol).
// ---------------------------------------------------------------------------
static __device__ __forceinline__ void
tdm_load_panel(const void* gbase, unsigned int lds_off, int rows,
               int row_dw, int stride_dw) {
  unsigned long long ga = (unsigned long long)(uintptr_t)gbase;
  u32x4 g0;
  g0[0] = 1u;                                            // count=1 (valid D#)
  g0[1] = lds_off;                                       // LDS byte address
  g0[2] = (unsigned int)(ga & 0xffffffffu);              // global addr [31:0]
  g0[3] = (unsigned int)((ga >> 32) & 0x01ffffffu)       // global addr [56:32]
        | (2u << 30);                                    // type = 2 (image)
  i32x8 g1;
  g1[0] = (int)(2u << 16);                               // data_size = 4B
  g1[1] = (int)(((unsigned)row_dw & 0xffffu) << 16);     // tensor_dim0 lo16
  g1[2] = (int)((((unsigned)row_dw >> 16) & 0xffffu)     // tensor_dim0 hi16
        | (((unsigned)rows & 0xffffu) << 16));           // tensor_dim1 lo16
  g1[3] = (int)(((unsigned)row_dw & 0xffffu) << 16);     // tile_dim0
  g1[4] = rows;                                          // tile_dim1 (tile_dim2=0)
  g1[5] = stride_dw;                                     // tensor_dim0_stride lo32
  g1[6] = 0;
  g1[7] = 0;
  i32x4 z4 = {0, 0, 0, 0};
  i32x8 z8 = {0, 0, 0, 0, 0, 0, 0, 0};
  __builtin_amdgcn_tensor_load_to_lds(g0, g1, z4, z4, z8, 0);
}

// ---------------------------------------------------------------------------
// P1: swizzle+convert weights [nmat x K x N] f32 -> fragment-native bf16
// ---------------------------------------------------------------------------
__global__ void k_swizzleB(const float* __restrict__ W, __bf16* __restrict__ out,
                           int K, int N, int KC, int nmat) {
  int id = blockIdx.x * blockDim.x + threadIdx.x;
  int NT = N >> 4;
  int per = KC * NT * 32;
  if (id >= per * nmat) return;
  int m = id / per, r = id - m * per;
  int lane = r & 31;
  int t = (r >> 5) % NT;
  int c = (r >> 5) / NT;
  int col = lane & 15;
  int kb = c * 32 + ((lane & 16) ? 16 : 0);
  const float* src = W + (size_t)m * K * N;
  __bf16 tmp[16];
#pragma unroll
  for (int j = 0; j < 16; ++j) {
    int k = kb + j;
    tmp[j] = (k < K) ? (__bf16)src[(size_t)k * N + t * 16 + col] : (__bf16)0.0f;
  }
  bf16x8 lo, hi;
#pragma unroll
  for (int j = 0; j < 8; ++j) { lo[j] = tmp[j]; hi[j] = tmp[8 + j]; }
  __bf16* dst = out + (size_t)id * 16;
  *(bf16x8*)dst = lo;
  *(bf16x8*)(dst + 8) = hi;
}

// ---------------------------------------------------------------------------
// P2: convert x [B,784] f32 -> bf16 padded to [B,800]
// ---------------------------------------------------------------------------
__global__ void k_padX(const float* __restrict__ x, __bf16* __restrict__ xbf) {
  size_t id = (size_t)blockIdx.x * blockDim.x + threadIdx.x;
  if (id >= (size_t)BATCH * KPAD) return;
  int k = (int)(id % KPAD);
  size_t b = id / KPAD;
  xbf[id] = (k < IN_DIM) ? (__bf16)x[b * IN_DIM + k] : (__bf16)0.0f;
}

// ---------------------------------------------------------------------------
// P3: convert f32 activations -> bf16 (4 elems/thread)
// ---------------------------------------------------------------------------
__global__ void k_cvt_act(const float* __restrict__ a, __bf16* __restrict__ o) {
  size_t id = ((size_t)blockIdx.x * blockDim.x + threadIdx.x) * 4;
  if (id >= (size_t)BATCH * HID) return;
  const float4 v = *(const float4*)(a + id);
  o[id + 0] = (__bf16)v.x; o[id + 1] = (__bf16)v.y;
  o[id + 2] = (__bf16)v.z; o[id + 3] = (__bf16)v.w;
}

// ---------------------------------------------------------------------------
// K1: proj = x @ W_in + b_in.  B panel (25KB) staged once per block via TDM.
// ---------------------------------------------------------------------------
__global__ void k_gemm_in(const __bf16* __restrict__ X, const __bf16* __restrict__ Wsw,
                          const float* __restrict__ bias, float* __restrict__ out) {
  __shared__ __align__(16) __bf16 bpanel[KC_IN * 512];
  int lane = threadIdx.x & 31, wave = threadIdx.x >> 5;
  int m0 = (blockIdx.x * 8 + wave) * 16;
  int nt = blockIdx.y;
  if (wave == 0) {
    tdm_load_panel(Wsw + (size_t)nt * 512,
                   (unsigned int)(uintptr_t)&bpanel[0], KC_IN, 256, 12288);
    __builtin_amdgcn_s_wait_tensorcnt(0);
  }
  __syncthreads();
  floatx8 acc = {};
  for (int c = 0; c < KC_IN; ++c) {
    bf16x16 a = load_a_bf(X, KPAD, m0, c * 32, lane);
    if (c + 1 < KC_IN)
      __builtin_prefetch(X + (size_t)(m0 + (lane & 15)) * KPAD + (c + 1) * 32, 0, 1);
    bf16x16 b = load_b_lds(bpanel, c, lane);
    acc = wmma_bf16(a, b, acc);
  }
  int col = lane & 15, mb = (lane & 16) ? 8 : 0;
  float bn = bias[nt * 16 + col];
#pragma unroll
  for (int i = 0; i < 8; ++i)
    out[(size_t)(m0 + mb + i) * HID + nt * 16 + col] = acc[i] + bn;
}

// ---------------------------------------------------------------------------
// K2: phasor bank features (one wave per row), emits bf16 feats directly
// ---------------------------------------------------------------------------
__global__ void k_phasor(const float* __restrict__ act, __bf16* __restrict__ feats) {
  int b = blockIdx.x, lane = threadIdx.x;
  const float* row = act + (size_t)b * HID;
  float s = 0.f;
  for (int j = lane; j < HID; j += 32) s += row[j];
#pragma unroll
  for (int off = 16; off; off >>= 1) s += __shfl_xor(s, off, 32);
  float mean = s * (1.0f / HID);
  float ph = mean * DELTA0 * (float)(lane + 1);
  feats[(size_t)b * 64 + lane]      = (__bf16)sinf(ph);
  feats[(size_t)b * 64 + 32 + lane] = (__bf16)cosf(ph);
}

// ---------------------------------------------------------------------------
// K3: enhanced = proj + feats @ W_ph + b_ph   (K=64, in-place accumulate)
// ---------------------------------------------------------------------------
__global__ void k_gemm_ph(const __bf16* __restrict__ F, const __bf16* __restrict__ Wsw,
                          const float* __restrict__ bias, float* __restrict__ act) {
  int lane = threadIdx.x & 31, wave = threadIdx.x >> 5;
  int m0 = (blockIdx.x * 8 + wave) * 16;
  int nt = blockIdx.y;
  floatx8 acc = {};
#pragma unroll
  for (int c = 0; c < 2; ++c) {
    bf16x16 a = load_a_bf(F, 64, m0, c * 32, lane);
    bf16x16 b = load_b_sw(Wsw, NT_HID, c, nt, lane);
    acc = wmma_bf16(a, b, acc);
  }
  int col = lane & 15, mb = (lane & 16) ? 8 : 0;
  float bn = bias[nt * 16 + col];
#pragma unroll
  for (int i = 0; i < 8; ++i) {
    size_t idx = (size_t)(m0 + mb + i) * HID + nt * 16 + col;
    act[idx] = act[idx] + acc[i] + bn;
  }
}

// ---------------------------------------------------------------------------
// K4: top-32 select -> decaying scatter -> 5-WTA gains; attended in place.
// ---------------------------------------------------------------------------
__global__ void k_attend(float* __restrict__ act) {
  __shared__ float buf[HID];
  int b = blockIdx.x, lane = threadIdx.x;
  float* row = act + (size_t)b * HID;
  for (int j = lane; j < HID; j += 32) buf[j] = row[j];
  __syncthreads();
  int myIdx = 0; float myW = 0.f;
  for (int i = 0; i < KSEL; ++i) {
    float mv = -3.0e38f; int mi = 0x7fffffff;
    for (int j = lane; j < HID; j += 32) {
      float v = buf[j];
      if (v > mv || (v == mv && j < mi)) { mv = v; mi = j; }
    }
#pragma unroll
    for (int off = 16; off; off >>= 1) {
      float ov = __shfl_xor(mv, off, 32);
      int   oi = __shfl_xor(mi, off, 32);
      if (ov > mv || (ov == mv && oi < mi)) { mv = ov; mi = oi; }
    }
    if (lane == i) { myIdx = mi; myW = powf(DECAY, (float)i); }
    if (lane == 0) buf[mi] = -3.0e38f;
    __syncthreads();
  }
  float v = myW;
  for (int t = 0; t < KWIN; ++t) {
    float bv = v; int bi = myIdx; int bl = lane;
#pragma unroll
    for (int off = 16; off; off >>= 1) {
      float ov = __shfl_xor(bv, off, 32);
      int   oi = __shfl_xor(bi, off, 32);
      int   ol = __shfl_xor(bl, off, 32);
      if (ov > bv || (ov == bv && oi < bi)) { bv = ov; bi = oi; bl = ol; }
    }
    if (lane == bl) v = -1.f;
    if (lane == 0) row[bi] *= (1.0f + bv);  // THETA == 1
  }
}

// ---------------------------------------------------------------------------
// K5: column mean of attended over batch (one block per column)
// ---------------------------------------------------------------------------
__global__ void k_colmean(const float* __restrict__ act, float* __restrict__ colmean) {
  __shared__ float red[256];
  int j = blockIdx.x, t = threadIdx.x;
  float s = 0.f;
  for (int b = t; b < BATCH; b += 256) s += act[(size_t)b * HID + j];
  red[t] = s; __syncthreads();
  for (int off = 128; off; off >>= 1) {
    if (t < off) red[t] += red[t + off];
    __syncthreads();
  }
  if (t == 0) colmean[j] = red[0] * (1.0f / BATCH);
}

// ---------------------------------------------------------------------------
// K6: bias8[e] = thal[e] + pbias[e]  (one wave per expert)
// ---------------------------------------------------------------------------
__global__ void k_bias8(const float* __restrict__ colmean, const float* __restrict__ W_th,
                        const float* __restrict__ b_th, const float* __restrict__ W_pm,
                        const float* __restrict__ b_pm, float* __restrict__ bias8) {
  int wave = threadIdx.x >> 5, lane = threadIdx.x & 31;
  if (wave >= NEXP) return;
  float s = 0.f;
  for (int j = lane; j < HID; j += 32) s += colmean[j] * W_th[(size_t)j * NEXP + wave];
#pragma unroll
  for (int off = 16; off; off >>= 1) s += __shfl_xor(s, off, 32);
  if (lane == 0) {
    const float tr[5] = {0.6f, 0.6f, 0.5f, 0.5f, 0.4f};
    float p = b_pm[wave];
#pragma unroll
    for (int t = 0; t < 5; ++t) p += tr[t] * W_pm[t * NEXP + wave];
    bias8[wave] = s + b_th[wave] + p;
  }
}

// ---------------------------------------------------------------------------
// K7: gate logits, top-2 softmax routing -> dense gate weights G[B,8]
// ---------------------------------------------------------------------------
__global__ void k_gate(const float* __restrict__ act, const float* __restrict__ W_gate,
                       const float* __restrict__ b_gate, const float* __restrict__ bias8,
                       float* __restrict__ G) {
  int b = blockIdx.x * blockDim.x + threadIdx.x;
  if (b >= BATCH) return;
  float acc[NEXP];
#pragma unroll
  for (int e = 0; e < NEXP; ++e) acc[e] = b_gate[e] + bias8[e];
  const float* row = act + (size_t)b * HID;
  for (int j = 0; j < HID; ++j) {
    float a = row[j];
#pragma unroll
    for (int e = 0; e < NEXP; ++e) acc[e] += a * W_gate[j * NEXP + e];
  }
  int e0 = 0;
#pragma unroll
  for (int e = 1; e < NEXP; ++e) if (acc[e] > acc[e0]) e0 = e;
  int e1 = (e0 == 0) ? 1 : 0;
#pragma unroll
  for (int e = 0; e < NEXP; ++e)
    if (e != e0 && acc[e] > acc[e1]) e1 = e;
  float m  = fmaxf(acc[e0], acc[e1]);
  float w0 = expf(acc[e0] - m), w1 = expf(acc[e1] - m);
  float inv = 1.0f / (w0 + w1);
#pragma unroll
  for (int e = 0; e < NEXP; ++e) G[(size_t)b * NEXP + e] = 0.f;
  G[(size_t)b * NEXP + e0] = w0 * inv;
  G[(size_t)b * NEXP + e1] = w1 * inv;
}

// ---------------------------------------------------------------------------
// K8: routed MoE combine. Per-expert B panel staged block-wide via TDM;
// per-expert WMMA accumulator, gate weight applied at the epilogue.
// Expert loop is block-uniform (barriers); absent experts skip only compute.
// ---------------------------------------------------------------------------
__global__ void k_experts(const __bf16* __restrict__ actbf, const float* __restrict__ G,
                          const __bf16* __restrict__ WexpSw, const float* __restrict__ b_exp,
                          float* __restrict__ ctx) {
  __shared__ __align__(16) __bf16 bpanel[KC_HID * 512];
  __shared__ float gt[8][16][NEXP];
  int lane = threadIdx.x & 31, wave = threadIdx.x >> 5;
  int m0 = (blockIdx.x * 8 + wave) * 16;
  int nt = blockIdx.y;
  if (lane < 16) {
#pragma unroll
    for (int e = 0; e < NEXP; ++e)
      gt[wave][lane][e] = G[(size_t)(m0 + lane) * NEXP + e];
  }
  int row = lane & 15, col = lane & 15, mb = (lane & 16) ? 8 : 0;
  float ctxacc[8];
#pragma unroll
  for (int i = 0; i < 8; ++i) ctxacc[i] = 0.f;

  for (int e = 0; e < NEXP; ++e) {
    __syncthreads();  // gt visible (e==0); previous panel reads done (e>0)
    if (wave == 0) {
      tdm_load_panel(WexpSw + ((size_t)e * KC_HID * NT_HID + nt) * 512,
                     (unsigned int)(uintptr_t)&bpanel[0], KC_HID, 256, 12288);
      __builtin_amdgcn_s_wait_tensorcnt(0);
    }
    __syncthreads();  // panel ready
    float gr = gt[wave][row][e];
    if (__ballot(gr != 0.0f) != 0ull) {
      floatx8 acc = {};
#pragma unroll 2
      for (int c = 0; c < KC_HID; ++c) {
        bf16x16 a = load_a_bf(actbf, HID, m0, c * 32, lane);
        bf16x16 b = load_b_lds(bpanel, c, lane);
        acc = wmma_bf16(a, b, acc);
      }
#pragma unroll
      for (int i = 0; i < 8; ++i) ctxacc[i] += gt[wave][mb + i][e] * acc[i];
    }
  }

  float be[NEXP];
#pragma unroll
  for (int e = 0; e < NEXP; ++e) be[e] = b_exp[e * HID + nt * 16 + col];
#pragma unroll
  for (int i = 0; i < 8; ++i) {
    int m = mb + i;
    float bias = 0.f;
#pragma unroll
    for (int e = 0; e < NEXP; ++e) bias += gt[wave][m][e] * be[e];
    ctx[(size_t)(m0 + m) * HID + nt * 16 + col] = ctxacc[i] + bias;
  }
}

// ---------------------------------------------------------------------------
// K9: out = context @ W_out + b_out   (N=10, memory bound, thread per row)
// ---------------------------------------------------------------------------
__global__ void k_out(const float* __restrict__ ctx, const float* __restrict__ W_out,
                      const float* __restrict__ b_out, float* __restrict__ out) {
  int b = blockIdx.x * blockDim.x + threadIdx.x;
  if (b >= BATCH) return;
  float acc[NCLS];
#pragma unroll
  for (int c = 0; c < NCLS; ++c) acc[c] = b_out[c];
  const float* row = ctx + (size_t)b * HID;
  for (int j = 0; j < HID; ++j) {
    float a = row[j];
#pragma unroll
    for (int c = 0; c < NCLS; ++c) acc[c] += a * W_out[j * NCLS + c];
  }
#pragma unroll
  for (int c = 0; c < NCLS; ++c) out[(size_t)b * NCLS + c] = acc[c];
}

// ---------------------------------------------------------------------------
extern "C" void kernel_launch(void* const* d_in, const int* in_sizes, int n_in,
                              void* d_out, int out_size, void* d_ws, size_t ws_size,
                              hipStream_t stream) {
  (void)in_sizes; (void)n_in; (void)out_size; (void)ws_size;
  const float* x      = (const float*)d_in[0];
  const float* W_in   = (const float*)d_in[1];
  const float* b_in   = (const float*)d_in[2];
  const float* W_ph   = (const float*)d_in[3];
  const float* b_ph   = (const float*)d_in[4];
  const float* W_gate = (const float*)d_in[5];
  const float* b_gate = (const float*)d_in[6];
  const float* W_exp  = (const float*)d_in[7];
  const float* b_exp  = (const float*)d_in[8];
  const float* W_th   = (const float*)d_in[9];
  const float* b_th   = (const float*)d_in[10];
  const float* W_pm   = (const float*)d_in[11];
  const float* b_pm   = (const float*)d_in[12];
  const float* W_out  = (const float*)d_in[13];
  const float* b_out  = (const float*)d_in[14];
  float* out = (float*)d_out;

  // ---- workspace carve (256B aligned chunks) ----
  char* wp = (char*)d_ws;
  auto carve = [&](size_t bytes) {
    char* p = wp; wp += (bytes + 255) & ~(size_t)255; return p;
  };
  float*   act     = (float*)  carve((size_t)BATCH * HID * 4);
  float*   ctx     = (float*)  carve((size_t)BATCH * HID * 4);
  __bf16*  Xbf     = (__bf16*) carve((size_t)BATCH * KPAD * 2);
  __bf16*  actbf   = (__bf16*) carve((size_t)BATCH * HID * 2);
  __bf16*  featsbf = (__bf16*) carve((size_t)BATCH * 64 * 2);
  __bf16*  WinSw   = (__bf16*) carve((size_t)KC_IN * NT_HID * 512 * 2);
  __bf16*  WphSw   = (__bf16*) carve((size_t)2 * NT_HID * 512 * 2);
  __bf16*  WexpSw  = (__bf16*) carve((size_t)NEXP * KC_HID * NT_HID * 512 * 2);
  float*   G       = (float*)  carve((size_t)BATCH * NEXP * 4);
  float*   colmean = (float*)  carve(HID * 4);
  float*   bias8   = (float*)  carve(NEXP * 4);

  // ---- weight pre-swizzle (f32 -> fragment-native bf16) ----
  {
    int tot = KC_IN * NT_HID * 32;
    k_swizzleB<<<(tot + 255) / 256, 256, 0, stream>>>(W_in, WinSw, IN_DIM, HID, KC_IN, 1);
  }
  {
    int tot = 2 * NT_HID * 32;
    k_swizzleB<<<(tot + 255) / 256, 256, 0, stream>>>(W_ph, WphSw, 64, HID, 2, 1);
  }
  {
    int tot = NEXP * KC_HID * NT_HID * 32;
    k_swizzleB<<<(tot + 255) / 256, 256, 0, stream>>>(W_exp, WexpSw, HID, HID, KC_HID, NEXP);
  }
  {
    size_t tot = (size_t)BATCH * KPAD;
    k_padX<<<(unsigned)((tot + 255) / 256), 256, 0, stream>>>(x, Xbf);
  }

  dim3 gGemm(BATCH / 16 / 8, NT_HID);  // (128, 48), 8 waves/block, tile/wave

  k_gemm_in<<<gGemm, 256, 0, stream>>>(Xbf, WinSw, b_in, act);
  k_phasor <<<BATCH, 32, 0, stream>>>(act, featsbf);
  k_gemm_ph<<<gGemm, 256, 0, stream>>>(featsbf, WphSw, b_ph, act);
  k_attend <<<BATCH, 32, 0, stream>>>(act);
  k_cvt_act<<<(BATCH * HID / 4 + 255) / 256, 256, 0, stream>>>(act, actbf);
  k_colmean<<<HID, 256, 0, stream>>>(act, colmean);
  k_bias8  <<<1, 256, 0, stream>>>(colmean, W_th, b_th, W_pm, b_pm, bias8);
  k_gate   <<<BATCH / 256, 256, 0, stream>>>(act, W_gate, b_gate, bias8, G);
  k_experts<<<gGemm, 256, 0, stream>>>(actbf, G, WexpSw, b_exp, ctx);
  k_out    <<<BATCH / 256, 256, 0, stream>>>(ctx, W_out, b_out, out);
}
